// ResonanceAttention_40673340293725
// MI455X (gfx1250) — matrix operand order repaired and checked by reading, not compiled, not run on hardware
//
#include <hip/hip_runtime.h>
#include <stdint.h>

#define D_MODEL 1024
#define N_HEADS 16
#define D_H     64
#define B_SZ    2
#define L_SEQ   2048
#define BL      (B_SZ * L_SEQ)     // 4096 rows
#define BHn     (B_SZ * N_HEADS)   // 32 (b,h) pairs
#define SROW    2052               // LDS row stride (floats), padded vs 2048
#define SMEM_BYTES ((16 * SROW + 256) * 4)   // 132352 B < 320 KB WGP LDS

typedef __attribute__((ext_vector_type(16))) __bf16 v16bf;
typedef __attribute__((ext_vector_type(8)))  float  v8f;

struct __attribute__((aligned(16))) U4 { unsigned int x, y, z, w; };

union Frag {
    U4             u4[2];
    unsigned short us[16];
    v16bf          v;
};

__device__ inline unsigned short f2bf(float f) {
    union { float f; unsigned int u; } cv; cv.f = f;
    unsigned int u = cv.u;
    return (unsigned short)((u + 0x7FFFu + ((u >> 16) & 1u)) >> 16);  // RNE
}

// A-fragment (16-bit 16x32): lane row = lo, k = hi*8 + e (+16 for e>=8)
__device__ inline Frag loadA(const unsigned short* __restrict__ row, int kb, int hi) {
    Frag A;
    A.u4[0] = *(const U4*)(row + kb + hi * 8);
    A.u4[1] = *(const U4*)(row + kb + 16 + hi * 8);
    return A;
}
// B-fragment (32x16): lane col = lo, k = hi*8 + e (+16 for e>=8)
__device__ inline Frag loadB(const unsigned short* __restrict__ row, int kb, int hi) {
    Frag B;
    B.u4[0] = *(const U4*)(row + kb + hi * 8);
    B.u4[1] = *(const U4*)(row + kb + 16 + hi * 8);
    return B;
}

__device__ inline v8f wmma_bf16(const Frag& a, const Frag& b, v8f c) {
    return __builtin_amdgcn_wmma_f32_16x16x32_bf16(false, a.v, false, b.v,
                                                   (short)0, c, false, false);
}

// ---------------------------------------------------------------- converts
__global__ __launch_bounds__(256) void cvt_f32_bf16(const float* __restrict__ in,
                                                    unsigned short* __restrict__ out, int n) {
    for (int i = blockIdx.x * 256 + threadIdx.x; i < n; i += gridDim.x * 256)
        out[i] = f2bf(in[i]);
}

// ------------------------------------------------- phase projection + tanh
__global__ __launch_bounds__(256) void phase_kernel(const float* __restrict__ x,
                                                    const float* __restrict__ phase_w,
                                                    const float* __restrict__ phase_b,
                                                    const float* __restrict__ pscale,
                                                    float* __restrict__ pv) {
    int idx = blockIdx.x * 256 + threadIdx.x;      // B*L*H = 65536
    int h  = idx & (N_HEADS - 1);
    int bl = idx >> 4;                              // b*L + l
    const float* xr = x + (size_t)bl * D_MODEL;
    const float* wr = phase_w + (size_t)h * D_MODEL;
    float acc = 0.f;
    for (int d = 0; d < D_MODEL; ++d) acc += xr[d] * wr[d];
    float t = tanhf(pscale[0] * (acc + phase_b[h]));
    int b = bl >> 11, l = bl & (L_SEQ - 1);
    pv[((b * N_HEADS + h) << 11) + l] = t;
}

// ---------------------------------------------------------------- QKV GEMM
__device__ inline void qkv_scatter(v8f c, int tm, int tn, int lo, int hi,
                                   const float* __restrict__ qkv_b,
                                   unsigned short* __restrict__ qbf,
                                   unsigned short* __restrict__ kbf,
                                   unsigned short* __restrict__ vT) {
    const int n = tn + lo;
    const int which = n >> 10, rem = n & 1023, h = rem >> 6, d = rem & 63;
    const float bias = qkv_b[n];
#pragma unroll
    for (int r = 0; r < 8; ++r) {
        int mm = tm + r + hi * 8;                   // global row in [0,4096)
        int b = mm >> 11, ls = mm & (L_SEQ - 1);
        int bh = b * N_HEADS + h;
        unsigned short bf = f2bf(c[r] + bias);
        if (which == 0)      qbf[(((size_t)bh << 11) + ls) * D_H + d] = bf;
        else if (which == 1) kbf[(((size_t)bh << 11) + ls) * D_H + d] = bf;
        else                 vT[(((size_t)bh * D_H + d) << 11) + ls]  = bf;
    }
}

// C[4096,3072] = Xbf @ Wbf^T + bias, 32x32 tile per wave (2x2 WMMA blocking)
__global__ __launch_bounds__(256) void qkv_gemm(const unsigned short* __restrict__ xbf,
                                                const unsigned short* __restrict__ wbf,
                                                const float* __restrict__ qkv_b,
                                                unsigned short* __restrict__ qbf,
                                                unsigned short* __restrict__ kbf,
                                                unsigned short* __restrict__ vT) {
    const int wave = (blockIdx.x * 256 + threadIdx.x) >> 5;
    const int lane = threadIdx.x & 31;
    const int lo = lane & 15, hi = lane >> 4;
    const int tn_tiles = 3 * D_MODEL / 32;          // 96
    const int tm = (wave / tn_tiles) * 32;
    const int tn = (wave % tn_tiles) * 32;
    const unsigned short* a0 = xbf + (size_t)(tm + lo) * D_MODEL;
    const unsigned short* a1 = a0 + (size_t)16 * D_MODEL;
    const unsigned short* b0 = wbf + (size_t)(tn + lo) * D_MODEL;
    const unsigned short* b1 = b0 + (size_t)16 * D_MODEL;
    v8f c00 = {}, c01 = {}, c10 = {}, c11 = {};
    for (int kb = 0; kb < D_MODEL; kb += 32) {
        Frag A0 = loadA(a0, kb, hi), A1 = loadA(a1, kb, hi);
        Frag B0 = loadB(b0, kb, hi), B1 = loadB(b1, kb, hi);
        c00 = wmma_bf16(A0, B0, c00);
        c01 = wmma_bf16(A0, B1, c01);
        c10 = wmma_bf16(A1, B0, c10);
        c11 = wmma_bf16(A1, B1, c11);
    }
    qkv_scatter(c00, tm,      tn,      lo, hi, qkv_b, qbf, kbf, vT);
    qkv_scatter(c01, tm,      tn + 16, lo, hi, qkv_b, qbf, kbf, vT);
    qkv_scatter(c10, tm + 16, tn,      lo, hi, qkv_b, qbf, kbf, vT);
    qkv_scatter(c11, tm + 16, tn + 16, lo, hi, qkv_b, qbf, kbf, vT);
}

// ---------------- fused scores + phase bias + softmax + attn@v -------------
// 1 block = 16 query rows of one (b,h). Scores strip (16 x 2048 fp32) lives
// in LDS; attn weights hit HBM exactly once (mandatory output write).
__global__ __launch_bounds__(256) void fused_attn(const unsigned short* __restrict__ qbf,
                                                  const unsigned short* __restrict__ kbf,
                                                  const unsigned short* __restrict__ vT,
                                                  const float* __restrict__ pv,
                                                  const float* __restrict__ rscale_p,
                                                  float* __restrict__ attnW,
                                                  unsigned short* __restrict__ aobf) {
    extern __shared__ float smem[];                 // [16][SROW] + red[256]
    float* red = smem + 16 * SROW;
    const int bh = blockIdx.x >> 7;                 // 128 row-blocks per (b,h)
    const int tm = (blockIdx.x & 127) * 16;
    const int tid = threadIdx.x;
    const int w = tid >> 5, lane = tid & 31;
    const int lo = lane & 15, hi = lane >> 4;
    const float rs = rscale_p[0];
    const float* pvr = pv + (size_t)bh * L_SEQ;

    // ---- phase 1: scores for rows [tm,tm+16) x cols [w*256, w*256+256)
    {
        const unsigned short* arow = qbf + ((size_t)bh * L_SEQ + tm + lo) * D_H;
        Frag A0 = loadA(arow, 0, hi);               // q rows reused for 16 tiles
        Frag A1 = loadA(arow, 32, hi);
#pragma unroll 4
        for (int tj = 0; tj < 16; ++tj) {
            const int tn = (w << 8) + (tj << 4);
            const unsigned short* brow = kbf + ((size_t)bh * L_SEQ + tn + lo) * D_H;
            Frag B0 = loadB(brow, 0, hi);
            Frag B1 = loadB(brow, 32, hi);
            v8f c = {};
            c = wmma_bf16(A0, B0, c);
            c = wmma_bf16(A1, B1, c);
            const float pn = pvr[tn + lo];
#pragma unroll
            for (int r = 0; r < 8; ++r) {
                int rl = r + hi * 8;
                float dp = pvr[tm + rl] - pn;
                smem[rl * SROW + tn + lo] = c[r] * 0.125f - rs * dp * dp;
            }
        }
    }
    __syncthreads();

    // ---- phase 2: softmax over 16 rows x 2048 in LDS; write weights out
    {
        const int row = tid >> 4, seg = tid & 15;   // 16 threads per row
        float* srow = smem + row * SROW;
        float mx = -3.402823466e38f;
#pragma unroll 8
        for (int j = 0; j < 128; ++j) mx = fmaxf(mx, srow[j * 16 + seg]);
        red[tid] = mx;
        __syncthreads();
#pragma unroll
        for (int i = 0; i < 16; ++i) mx = fmaxf(mx, red[(row << 4) + i]);
        float sm = 0.f;
#pragma unroll 8
        for (int j = 0; j < 128; ++j) {
            float e = __expf(srow[j * 16 + seg] - mx);
            srow[j * 16 + seg] = e;
            sm += e;
        }
        __syncthreads();                            // all reads of red done
        red[tid] = sm;
        __syncthreads();
        float tot = 0.f;
#pragma unroll
        for (int i = 0; i < 16; ++i) tot += red[(row << 4) + i];
        const float inv = 1.0f / tot;
        float* grow = attnW + ((size_t)bh * L_SEQ + tm + row) * L_SEQ;
#pragma unroll 8
        for (int j = 0; j < 128; ++j) {
            float v = srow[j * 16 + seg] * inv;
            srow[j * 16 + seg] = v;
            grow[j * 16 + seg] = v;                 // the only HBM pass
        }
    }
    __syncthreads();

    // ---- phase 3: out[16 x 64] = weights(LDS) @ v ; 4 waves, 1 tile each
    if (w < 4) {
        const int tn = w << 4;                      // dh column tile
        const unsigned short* brow = vT + ((size_t)bh * D_H + tn + lo) * L_SEQ;
        const float* aLds = smem + lo * SROW;       // lane row = lo
        v8f c = {};
        for (int kb = 0; kb < L_SEQ; kb += 32) {
            Frag A;
            const float* p0 = aLds + kb + hi * 8;
            const float* p1 = aLds + kb + 16 + hi * 8;
#pragma unroll
            for (int e = 0; e < 8; ++e) A.us[e] = f2bf(p0[e]);
#pragma unroll
            for (int e = 0; e < 8; ++e) A.us[8 + e] = f2bf(p1[e]);
            Frag B = loadB(brow, kb, hi);
            c = wmma_bf16(A, B, c);
        }
        const int b = bh >> 4, h = bh & (N_HEADS - 1);
        const int col = h * D_H + tn + lo;
#pragma unroll
        for (int r = 0; r < 8; ++r) {
            int mm = tm + r + hi * 8;
            aobf[(size_t)(b * L_SEQ + mm) * D_MODEL + col] = f2bf(c[r]);
        }
    }
}

// -------------------------------------------------------- output projection
__global__ __launch_bounds__(256) void out_gemm(const unsigned short* __restrict__ aobf,
                                                const unsigned short* __restrict__ wbf,
                                                const float* __restrict__ out_b,
                                                float* __restrict__ out) {
    const int wave = (blockIdx.x * 256 + threadIdx.x) >> 5;
    const int lane = threadIdx.x & 31;
    const int lo = lane & 15, hi = lane >> 4;
    const int tm = (wave >> 5) * 32;                 // 32 n-wave-tiles
    const int tn = (wave & 31) * 32;
    const unsigned short* a0 = aobf + (size_t)(tm + lo) * D_MODEL;
    const unsigned short* a1 = a0 + (size_t)16 * D_MODEL;
    const unsigned short* b0 = wbf + (size_t)(tn + lo) * D_MODEL;
    const unsigned short* b1 = b0 + (size_t)16 * D_MODEL;
    v8f c00 = {}, c01 = {}, c10 = {}, c11 = {};
    for (int kb = 0; kb < D_MODEL; kb += 32) {
        Frag A0 = loadA(a0, kb, hi), A1 = loadA(a1, kb, hi);
        Frag B0 = loadB(b0, kb, hi), B1 = loadB(b1, kb, hi);
        c00 = wmma_bf16(A0, B0, c00);
        c01 = wmma_bf16(A0, B1, c01);
        c10 = wmma_bf16(A1, B0, c10);
        c11 = wmma_bf16(A1, B1, c11);
    }
#pragma unroll
    for (int i = 0; i < 4; ++i) {
        v8f c = (i == 0) ? c00 : (i == 1) ? c01 : (i == 2) ? c10 : c11;
        int stm = tm + ((i >> 1) ? 16 : 0);
        int n   = tn + ((i & 1) ? 16 : 0) + lo;
        float bias = out_b[n];
#pragma unroll
        for (int r = 0; r < 8; ++r) {
            int mm = stm + r + hi * 8;
            out[(size_t)mm * D_MODEL + n] = c[r] + bias;
        }
    }
}

// ---------------------------------------------------------------- launcher
extern "C" void kernel_launch(void* const* d_in, const int* in_sizes, int n_in,
                              void* d_out, int out_size, void* d_ws, size_t ws_size,
                              hipStream_t stream) {
    const float* x       = (const float*)d_in[0];
    const float* qkv_w   = (const float*)d_in[1];
    const float* qkv_b   = (const float*)d_in[2];
    const float* out_w   = (const float*)d_in[3];
    const float* out_b   = (const float*)d_in[4];
    const float* phase_w = (const float*)d_in[5];
    const float* phase_b = (const float*)d_in[6];
    const float* rscale  = (const float*)d_in[7];
    const float* pscale  = (const float*)d_in[8];

    char* ws = (char*)d_ws;   // ~48.3 MB used, all offsets 16B-aligned
    unsigned short* xbf   = (unsigned short*)(ws + 0);          //  8 MB
    unsigned short* qkvwb = (unsigned short*)(ws + 8388608);    //  6 MB
    unsigned short* outwb = (unsigned short*)(ws + 14680064);   //  2 MB
    unsigned short* qbf   = (unsigned short*)(ws + 16777216);   //  8 MB
    unsigned short* kbf   = (unsigned short*)(ws + 25165824);   //  8 MB
    unsigned short* vT    = (unsigned short*)(ws + 33554432);   //  8 MB
    float*          pv    = (float*)(ws + 41943040);            // 256 KB
    unsigned short* aobf  = (unsigned short*)(ws + 42205184);   //  8 MB

    float* out   = (float*)d_out;
    float* attnW = out + (size_t)BL * D_MODEL;   // attn_weights after out

    cvt_f32_bf16<<<4096, 256, 0, stream>>>(x, xbf, BL * D_MODEL);
    cvt_f32_bf16<<<4096, 256, 0, stream>>>(qkv_w, qkvwb, 3 * D_MODEL * D_MODEL);
    cvt_f32_bf16<<<2048, 256, 0, stream>>>(out_w, outwb, D_MODEL * D_MODEL);

    phase_kernel<<<(BL * N_HEADS) / 256, 256, 0, stream>>>(x, phase_w, phase_b, pscale, pv);

    // 128 x 96 wave-tiles (32x32 each), 8 waves/block
    qkv_gemm<<<(BL / 32) * (3 * D_MODEL / 32) / 8, 256, 0, stream>>>(
        xbf, qkvwb, qkv_b, qbf, kbf, vT);

    // fused scores+softmax+attn@v: 32 (b,h) x 128 row-blocks, 132 KB LDS each
    fused_attn<<<BHn * (L_SEQ / 16), 256, SMEM_BYTES, stream>>>(
        qbf, kbf, vT, pv, rscale, attnW, aobf);

    // 128 * 32 wave-tiles
    out_gemm<<<(BL / 32) * (D_MODEL / 32) / 8, 256, 0, stream>>>(aobf, outwb, out_b, out);
}
